// EnhancedPerceptionModule_48808008351772
// MI455X (gfx1250) — compile-verified, matrix operand order checked by reading, not run on hardware
//
#include <hip/hip_runtime.h>
#include <hip/hip_bf16.h>
#include <math.h>

// ---------------------------------------------------------------------------
// EnhancedPerceptionModule for MI455X (gfx1250, wave32, WMMA).
// GEMM-shaped ops (projections, attention, gating, routing logits, heads,
// relation MLP, and the 3x3 FPN convs via im2col) all use
// v_wmma_f32_16x16x32_f16 (f16 in, f32 accum) with LDS-staged weight tiles,
// 4x register blocking, and branch-free global loads.
// ---------------------------------------------------------------------------

#define FDIM 512
#define NCON 100
#define NHEAD 8
#define DHEAD 64
#define NB 4
#define CK 2304                       // 3x3 conv im2col depth: 256*9

typedef __attribute__((ext_vector_type(16))) _Float16 v16h;
typedef __attribute__((ext_vector_type(8)))  float    v8f;

union AFrag { v16h v; _Float16 h[16]; };
union CFrag { v8f  v; float     f[8];  };

// K-offset (within a 32-wide K step) for element e of a 16-bit A/B fragment.
// ISA 7.12.2: lanes 0-15 hold K {0..7, 16..23}; lanes 16-31 hold K {8..15,
// 24..31}.  B assumed symmetric with N on (lane&15).
__device__ __forceinline__ int kmap16(int e, int lane) {
    return (e & 7) + ((e >> 3) << 4) + ((lane >> 4) << 3);
}

__device__ __forceinline__ int imin_(int a, int b) { return a < b ? a : b; }
__device__ __forceinline__ int imax_(int a, int b) { return a > b ? a : b; }

#define ACT_NONE 0
#define ACT_RELU 1
#define ACT_GELU 2
#define ACT_SIGM 3

__device__ __forceinline__ float act_apply(float y, int act) {
    if (act == ACT_RELU) return fmaxf(y, 0.f);
    if (act == ACT_GELU) return 0.5f * y * (1.f + erff(y * 0.70710678118654752f));
    if (act == ACT_SIGM) return 1.f / (1.f + expf(-y));
    return y;
}

// Pack a float4 pair (8 consecutive K values) into fragment halves.
__device__ __forceinline__ void cvt8(AFrag& fr, int base, const float4& a, const float4& b) {
    fr.h[base + 0] = (_Float16)a.x; fr.h[base + 1] = (_Float16)a.y;
    fr.h[base + 2] = (_Float16)a.z; fr.h[base + 3] = (_Float16)a.w;
    fr.h[base + 4] = (_Float16)b.x; fr.h[base + 5] = (_Float16)b.y;
    fr.h[base + 6] = (_Float16)b.z; fr.h[base + 7] = (_Float16)b.w;
}

// LDS B-tile pitch: 40 halves = 80B -> 16B aligned rows, conflict-spread.
#define BPITCH 40

// ---------------------------------------------------------------------------
// Generic GEMM: C[M,N] = act(A[M,K] @ W[K,N] + bias), row-major.
// Block = 128 threads (4 waves).  Block tile = 64(M) x 64(N); each wave owns
// 16 rows x 64 cols = 4 accumulators, sharing one LDS-staged 32x64 W tile.
// ---------------------------------------------------------------------------
__global__ void gemm_wmma(const float* __restrict__ A, const float* __restrict__ W,
                          const float* __restrict__ bias, float* __restrict__ C,
                          int M, int N, int K, int ldc, int act)
{
    __shared__ _Float16 ldsB[64][BPITCH];       // [n][k], transposed tile
    const int tid  = threadIdx.x;
    const int lane = tid & 31;
    const int wave = tid >> 5;
    const int m0 = blockIdx.x * 64 + wave * 16;
    const int n0 = blockIdx.y * 64;

    const int arow = m0 + (lane & 15);
    const float* Arow = A + (size_t)imin_(arow, M - 1) * K;   // clamped: safe, discarded
    const int khi = (lane >> 4) << 3;                          // 0 or 8

    // staging role: thread t covers W row k = t&31, 16-col group sq = (t>>5)*16
    const int sk = tid & 31;
    const int sq = (tid >> 5) << 4;

    CFrag acc[4];
#pragma unroll
    for (int nt = 0; nt < 4; ++nt)
#pragma unroll
        for (int i = 0; i < 8; ++i) acc[nt].f[i] = 0.f;

    for (int k0 = 0; k0 < K; k0 += 32) {
        // ---- stage B tile (32K x 64N) -> LDS (f16, transposed) ----
        {
            const int kk = k0 + sk;
            const float* Wr = W + (size_t)imin_(kk, K - 1) * N;
            if (n0 + 64 <= N && k0 + 32 <= K) {           // uniform fast path
#pragma unroll
                for (int f = 0; f < 4; ++f) {
                    const float4 v = *(const float4*)(Wr + n0 + sq + 4 * f);
                    ldsB[sq + 4 * f + 0][sk] = (_Float16)v.x;
                    ldsB[sq + 4 * f + 1][sk] = (_Float16)v.y;
                    ldsB[sq + 4 * f + 2][sk] = (_Float16)v.z;
                    ldsB[sq + 4 * f + 3][sk] = (_Float16)v.w;
                }
            } else {                                      // uniform tail path
#pragma unroll
                for (int c = 0; c < 16; ++c) {
                    const int col = n0 + sq + c;
                    float v = Wr[imin_(col, N - 1)];
                    v = (col < N && kk < K) ? v : 0.f;
                    ldsB[sq + c][sk] = (_Float16)v;
                }
            }
        }
        __syncthreads();

        // ---- A fragment: two contiguous 8-float runs, vector loads ----
        AFrag af;
        if (k0 + 32 <= K) {
            __builtin_prefetch(Arow + k0 + 64, 0, 1);
            const float4 a0 = *(const float4*)(Arow + k0 + khi);
            const float4 a1 = *(const float4*)(Arow + k0 + khi + 4);
            const float4 a2 = *(const float4*)(Arow + k0 + khi + 16);
            const float4 a3 = *(const float4*)(Arow + k0 + khi + 20);
            cvt8(af, 0, a0, a1);
            cvt8(af, 8, a2, a3);
        } else {
#pragma unroll
            for (int e = 0; e < 16; ++e) {
                const int kk = k0 + kmap16(e, lane);
                float v = Arow[imin_(kk, K - 1)];
                af.h[e] = (_Float16)((kk < K) ? v : 0.f);
            }
        }

        // ---- 4 B fragments from LDS + 4 back-to-back WMMAs ----
#pragma unroll
        for (int nt = 0; nt < 4; ++nt) {
            AFrag bf;
            const _Float16* bp = &ldsB[nt * 16 + (lane & 15)][khi];
#pragma unroll
            for (int e = 0; e < 8; ++e) { bf.h[e] = bp[e]; bf.h[8 + e] = bp[16 + e]; }
            acc[nt].v = __builtin_amdgcn_wmma_f32_16x16x32_f16(false, af.v, false, bf.v,
                                                               (short)0, acc[nt].v, false, false);
        }
        __syncthreads();
    }

    // ---- epilogue ----
    const int rbase = m0 + ((lane >> 4) << 3);
#pragma unroll
    for (int nt = 0; nt < 4; ++nt) {
        const int col = n0 + nt * 16 + (lane & 15);
        if (col >= N) continue;
        const float bv = bias ? bias[col] : 0.f;
#pragma unroll
        for (int v = 0; v < 8; ++v) {
            const int row = rbase + v;
            if (row < M) C[(size_t)row * ldc + col] = act_apply(acc[nt].f[v] + bv, act);
        }
    }
}

// ---------------------------------------------------------------------------
// Attention scores: S[bh,n,m] = scale * sum_d Q[b,n,h*64+d]*K[b,m,h*64+d]
// K = 64 -> both fragments are pure float4 loads; rows clamped.
// grid = (ceil(nq/16), ceil(nkv/16), NB*NHEAD), block = 32.
// ---------------------------------------------------------------------------
__global__ void attn_qk_wmma(const float* __restrict__ Q, const float* __restrict__ Kp,
                             float* __restrict__ S, int nq, int nkv, float scale)
{
    const int bh = blockIdx.z, b = bh / NHEAD, h = bh % NHEAD;
    const int lane = threadIdx.x & 31;
    const int n0 = blockIdx.x * 16, m0 = blockIdx.y * 16;
    const int qrow = n0 + (lane & 15);
    const int krow = m0 + (lane & 15);
    const int khi = (lane >> 4) << 3;
    const float* Qr = Q + ((size_t)b * nq + imin_(qrow, nq - 1)) * FDIM + h * DHEAD;
    const float* Kr = Kp + ((size_t)b * nkv + imin_(krow, nkv - 1)) * FDIM + h * DHEAD;

    CFrag acc;
#pragma unroll
    for (int i = 0; i < 8; ++i) acc.f[i] = 0.f;

#pragma unroll
    for (int k0 = 0; k0 < DHEAD; k0 += 32) {
        AFrag af, bf;
        cvt8(af, 0, *(const float4*)(Qr + k0 + khi),      *(const float4*)(Qr + k0 + khi + 4));
        cvt8(af, 8, *(const float4*)(Qr + k0 + khi + 16), *(const float4*)(Qr + k0 + khi + 20));
        cvt8(bf, 0, *(const float4*)(Kr + k0 + khi),      *(const float4*)(Kr + k0 + khi + 4));
        cvt8(bf, 8, *(const float4*)(Kr + k0 + khi + 16), *(const float4*)(Kr + k0 + khi + 20));
        acc.v = __builtin_amdgcn_wmma_f32_16x16x32_f16(false, af.v, false, bf.v,
                                                       (short)0, acc.v, false, false);
    }
    const int col = m0 + (lane & 15);
    if (col >= nkv) return;
    const int rbase = n0 + ((lane >> 4) << 3);
#pragma unroll
    for (int v = 0; v < 8; ++v) {
        const int row = rbase + v;
        if (row < nq) S[((size_t)bh * nq + row) * nkv + col] = acc.f[v] * scale;
    }
}

// ---------------------------------------------------------------------------
// Attention output: O[b,n,h*64+d] = sum_m S[bh,n,m] * V[b,m,h*64+d]
// Block = 128 threads; block tile = 64(n) x 64(d) (full head dim).
// V tile (32 x 64) staged in LDS and shared by 4 waves.
// grid = (ceil(nq/64), NB*NHEAD), block = 128.
// ---------------------------------------------------------------------------
__global__ void attn_av_wmma(const float* __restrict__ S, const float* __restrict__ V,
                             float* __restrict__ O, int nq, int nkv)
{
    __shared__ _Float16 ldsB[64][BPITCH];
    const int bh = blockIdx.y, b = bh / NHEAD, h = bh % NHEAD;
    const int tid = threadIdx.x;
    const int lane = tid & 31;
    const int wave = tid >> 5;
    const int m0 = blockIdx.x * 64 + wave * 16;
    const int arow = m0 + (lane & 15);
    const float* Srow = S + ((size_t)bh * nq + imin_(arow, nq - 1)) * nkv;
    const float* Vb = V + (size_t)b * nkv * FDIM + h * DHEAD;
    const int khi = (lane >> 4) << 3;
    const int sk = tid & 31;
    const int sq = (tid >> 5) << 4;

    CFrag acc[4];
#pragma unroll
    for (int nt = 0; nt < 4; ++nt)
#pragma unroll
        for (int i = 0; i < 8; ++i) acc[nt].f[i] = 0.f;

    for (int k0 = 0; k0 < nkv; k0 += 32) {
        // stage V tile (32 x 64) -> LDS transposed
        {
            const int kk = k0 + sk;
            const float* Vr = Vb + (size_t)imin_(kk, nkv - 1) * FDIM;
            if (k0 + 32 <= nkv) {
#pragma unroll
                for (int f = 0; f < 4; ++f) {
                    const float4 v = *(const float4*)(Vr + sq + 4 * f);
                    ldsB[sq + 4 * f + 0][sk] = (_Float16)v.x;
                    ldsB[sq + 4 * f + 1][sk] = (_Float16)v.y;
                    ldsB[sq + 4 * f + 2][sk] = (_Float16)v.z;
                    ldsB[sq + 4 * f + 3][sk] = (_Float16)v.w;
                }
            } else {
#pragma unroll
                for (int c = 0; c < 16; ++c) {
                    float v = Vr[sq + c];
                    ldsB[sq + c][sk] = (_Float16)((kk < nkv) ? v : 0.f);
                }
            }
        }
        __syncthreads();

        AFrag af;
        if (k0 + 32 <= nkv) {
            cvt8(af, 0, *(const float4*)(Srow + k0 + khi),      *(const float4*)(Srow + k0 + khi + 4));
            cvt8(af, 8, *(const float4*)(Srow + k0 + khi + 16), *(const float4*)(Srow + k0 + khi + 20));
        } else {
#pragma unroll
            for (int e = 0; e < 16; ++e) {
                const int kk = k0 + kmap16(e, lane);
                float v = Srow[imin_(kk, nkv - 1)];
                af.h[e] = (_Float16)((kk < nkv) ? v : 0.f);
            }
        }

#pragma unroll
        for (int nt = 0; nt < 4; ++nt) {
            AFrag bf;
            const _Float16* bp = &ldsB[nt * 16 + (lane & 15)][khi];
#pragma unroll
            for (int e = 0; e < 8; ++e) { bf.h[e] = bp[e]; bf.h[8 + e] = bp[16 + e]; }
            acc[nt].v = __builtin_amdgcn_wmma_f32_16x16x32_f16(false, af.v, false, bf.v,
                                                               (short)0, acc[nt].v, false, false);
        }
        __syncthreads();
    }

    const int rbase = m0 + ((lane >> 4) << 3);
#pragma unroll
    for (int nt = 0; nt < 4; ++nt) {
        const int col = nt * 16 + (lane & 15);           // < 64 always
#pragma unroll
        for (int v = 0; v < 8; ++v) {
            const int row = rbase + v;
            if (row < nq)
                O[((size_t)b * nq + row) * FDIM + h * DHEAD + col] = acc[nt].f[v];
        }
    }
}

// ---------------------------------------------------------------------------
// Relation layer 1 (fused gather-GEMM):
//   h1[m,n] = relu([cf[b,i] | cf[b,j]] @ W1[0:1024] + pd @ W1[1024:1026] + b1)
// m = (b*100+i)*100+j; M = 40000 (mult of 64), N = 512, K = 1024 -> no guards.
// A runs never straddle the 512 boundary (k0 is a multiple of 32), so the
// cfi/cfj side select is uniform per K step.
// grid = (625, 8), block = 128.
// ---------------------------------------------------------------------------
__global__ void rel1_wmma(const float* __restrict__ cf, const float* __restrict__ pos,
                          const float* __restrict__ W1, const float* __restrict__ b1,
                          float* __restrict__ h1)
{
    __shared__ _Float16 ldsB[64][BPITCH];
    const int tid = threadIdx.x;
    const int lane = tid & 31;
    const int wave = tid >> 5;
    const int m0 = blockIdx.x * 64 + wave * 16;
    const int n0 = blockIdx.y * 64;

    const int arow = m0 + (lane & 15);
    const int b = arow / (NCON * NCON);
    const int ij = arow % (NCON * NCON);
    const int i = ij / NCON, j = ij % NCON;
    const float* cfi = cf + ((size_t)b * NCON + i) * FDIM;
    const float* cfj = cf + ((size_t)b * NCON + j) * FDIM;
    const int khi = (lane >> 4) << 3;
    const int sk = tid & 31;
    const int sq = (tid >> 5) << 4;

    CFrag acc[4];
#pragma unroll
    for (int nt = 0; nt < 4; ++nt)
#pragma unroll
        for (int v = 0; v < 8; ++v) acc[nt].f[v] = 0.f;

    for (int k0 = 0; k0 < 2 * FDIM; k0 += 32) {
        {
            const float* Wr = W1 + (size_t)(k0 + sk) * FDIM;
#pragma unroll
            for (int f = 0; f < 4; ++f) {
                const float4 v = *(const float4*)(Wr + n0 + sq + 4 * f);
                ldsB[sq + 4 * f + 0][sk] = (_Float16)v.x;
                ldsB[sq + 4 * f + 1][sk] = (_Float16)v.y;
                ldsB[sq + 4 * f + 2][sk] = (_Float16)v.z;
                ldsB[sq + 4 * f + 3][sk] = (_Float16)v.w;
            }
        }
        __syncthreads();

        const float* src = (k0 < FDIM) ? cfi : cfj;      // uniform per K step
        const int kb = (k0 < FDIM) ? k0 : k0 - FDIM;
        AFrag af;
        cvt8(af, 0, *(const float4*)(src + kb + khi),      *(const float4*)(src + kb + khi + 4));
        cvt8(af, 8, *(const float4*)(src + kb + khi + 16), *(const float4*)(src + kb + khi + 20));

#pragma unroll
        for (int nt = 0; nt < 4; ++nt) {
            AFrag bf;
            const _Float16* bp = &ldsB[nt * 16 + (lane & 15)][khi];
#pragma unroll
            for (int e = 0; e < 8; ++e) { bf.h[e] = bp[e]; bf.h[8 + e] = bp[16 + e]; }
            acc[nt].v = __builtin_amdgcn_wmma_f32_16x16x32_f16(false, af.v, false, bf.v,
                                                               (short)0, acc[nt].v, false, false);
        }
        __syncthreads();
    }

    // epilogue: position-delta columns (K = 1024, 1025) + bias + relu
    const int rbase = m0 + ((lane >> 4) << 3);
#pragma unroll
    for (int nt = 0; nt < 4; ++nt) {
        const int col = n0 + nt * 16 + (lane & 15);
        const float w0 = W1[(size_t)1024 * FDIM + col];
        const float w1 = W1[(size_t)1025 * FDIM + col];
        const float bv = b1[col];
#pragma unroll
        for (int v = 0; v < 8; ++v) {
            const int row = rbase + v;
            const int bb = row / (NCON * NCON);
            const int ij2 = row % (NCON * NCON);
            const int ii = ij2 / NCON, jj = ij2 % NCON;
            const float pdx = pos[(bb * NCON + ii) * 2 + 0] - pos[(bb * NCON + jj) * 2 + 0];
            const float pdy = pos[(bb * NCON + ii) * 2 + 1] - pos[(bb * NCON + jj) * 2 + 1];
            const float y = acc[nt].f[v] + pdx * w0 + pdy * w1 + bv;
            h1[(size_t)row * FDIM + col] = fmaxf(y, 0.f);
        }
    }
}

// ------------------------- scalar glue kernels -----------------------------

__global__ void conv1x1(const float* __restrict__ in, const float* __restrict__ w,
                        const float* __restrict__ bias, float* __restrict__ out,
                        int Cin, int HW)
{
    const int idx = blockIdx.x * blockDim.x + threadIdx.x;
    const int total = NB * 256 * HW;
    if (idx >= total) return;
    const int s = idx % HW;
    const int o = (idx / HW) % 256;
    const int b = idx / (HW * 256);
    const float* ip = in + (size_t)b * Cin * HW + s;
    const float* wp = w + (size_t)o * Cin;
    float acc = bias[o];
    for (int c = 0; c < Cin; ++c) acc += wp[c] * ip[(size_t)c * HW];
    out[idx] = acc;
}

__global__ void upsample_add(const float* __restrict__ lat, const float* __restrict__ src,
                             float* __restrict__ out, int H)
{
    const int idx = blockIdx.x * blockDim.x + threadIdx.x;
    const int total = NB * 256 * H * H;
    if (idx >= total) return;
    const int x = idx % H;
    const int y = (idx / H) % H;
    const int c = (idx / (H * H)) % 256;
    const int b = idx / (H * H * 256);
    const int hh = H >> 1;
    out[idx] = lat[idx] + src[(((size_t)b * 256 + c) * hh + (y >> 1)) * hh + (x >> 1)];
}

// im2col for 3x3 pad-1 conv (NCHW input):
// col[(b*HW + s)*CK + c*9 + ky*3 + kx] = in[b,c,y+ky-1,x+kx-1] (0 outside)
__global__ void im2col3x3(const float* __restrict__ in, float* __restrict__ col, int H)
{
    const int HW = H * H;
    const long total = (long)NB * HW * CK;
    const long idx = (long)blockIdx.x * blockDim.x + threadIdx.x;
    if (idx >= total) return;
    const int kk = (int)(idx % 9);
    const int c  = (int)((idx / 9) % 256);
    const int s  = (int)((idx / CK) % HW);
    const int b  = (int)(idx / ((long)CK * HW));
    const int y = s / H, x = s % H;
    const int yy = y + (kk / 3) - 1;
    const int xx = x + (kk % 3) - 1;
    float v = 0.f;
    if (yy >= 0 && yy < H && xx >= 0 && xx < H)
        v = in[((size_t)b * 256 + c) * HW + yy * H + xx];
    col[idx] = v;
}

// transpose conv weights OIHW [O][K] -> [K][O] for the GEMM B operand
__global__ void transpose_oc(const float* __restrict__ w, float* __restrict__ wt, int O, int K)
{
    const int idx = blockIdx.x * blockDim.x + threadIdx.x;
    if (idx >= O * K) return;
    const int o = idx % O;
    const int k = idx / O;
    wt[idx] = w[(size_t)o * K + k];     // wt[k*O + o]
}

// bilinear sample of a token-major [(y*H+x)*256 + c] map at 28x28 grid coords
__device__ __forceinline__ float bilin_t(const float* f, int H, int y, int x, int c)
{
    const float scale = (float)H / 28.f;
    const float sy = ((float)y + 0.5f) * scale - 0.5f;
    const float sx = ((float)x + 0.5f) * scale - 0.5f;
    int y0 = (int)floorf(sy), x0 = (int)floorf(sx);
    const float wy = sy - (float)y0, wx = sx - (float)x0;
    int y1 = y0 + 1, x1 = x0 + 1;
    y0 = imin_(imax_(y0, 0), H - 1); y1 = imin_(imax_(y1, 0), H - 1);
    x0 = imin_(imax_(x0, 0), H - 1); x1 = imin_(imax_(x1, 0), H - 1);
    const float s00 = f[((size_t)(y0 * H + x0)) * 256 + c];
    const float s01 = f[((size_t)(y0 * H + x1)) * 256 + c];
    const float s10 = f[((size_t)(y1 * H + x0)) * 256 + c];
    const float s11 = f[((size_t)(y1 * H + x1)) * 256 + c];
    return (1.f - wy) * ((1.f - wx) * s00 + wx * s01) + wy * ((1.f - wx) * s10 + wx * s11);
}

// tokens: x0[b,t,c] = mean(c0, bilinear(c1), bilinear(c2)) + PE(t, c)
// conv outputs are token-major: cX[(b*HW + s)*256 + c]
__global__ void build_tokens(const float* __restrict__ c0, const float* __restrict__ c1,
                             const float* __restrict__ c2, float* __restrict__ xout)
{
    const int idx = blockIdx.x * blockDim.x + threadIdx.x;
    const int total = NB * 784 * 256;
    if (idx >= total) return;
    const int c = idx % 256;
    const int t = (idx / 256) % 784;
    const int b = idx / (256 * 784);
    const int y = t / 28, x = t % 28;
    const float* f0 = c0 + (size_t)b * 784 * 256;
    const float* f1 = c1 + (size_t)b * 196 * 256;
    const float* f2 = c2 + (size_t)b * 49 * 256;
    float v = (f0[(size_t)t * 256 + c] + bilin_t(f1, 14, y, x, c) + bilin_t(f2, 7, y, x, c))
              * (1.f / 3.f);
    const int jj = c >> 1;
    const float ang = (float)t * expf((float)(2 * jj) * (-9.210340371976184f / 256.f));
    v += (c & 1) ? cosf(ang) : sinf(ang);
    xout[idx] = v;
}

// per-row layernorm (dim 512) followed by exact GELU
__global__ void ln_gelu(const float* __restrict__ X, const float* __restrict__ g,
                        const float* __restrict__ bta, float* __restrict__ Y)
{
    __shared__ float red[256];
    const int tid = threadIdx.x;
    const float* x = X + (size_t)blockIdx.x * FDIM;
    float* y = Y + (size_t)blockIdx.x * FDIM;
    float s = 0.f;
    for (int i = tid; i < FDIM; i += 256) s += x[i];
    red[tid] = s; __syncthreads();
    for (int st = 128; st > 0; st >>= 1) { if (tid < st) red[tid] += red[tid + st]; __syncthreads(); }
    const float mean = red[0] * (1.f / FDIM);
    __syncthreads();
    float v = 0.f;
    for (int i = tid; i < FDIM; i += 256) { const float d = x[i] - mean; v += d * d; }
    red[tid] = v; __syncthreads();
    for (int st = 128; st > 0; st >>= 1) { if (tid < st) red[tid] += red[tid + st]; __syncthreads(); }
    const float inv = rsqrtf(red[0] * (1.f / FDIM) + 1e-5f);
    for (int i = tid; i < FDIM; i += 256) {
        const float t = (x[i] - mean) * inv * g[i] + bta[i];
        y[i] = 0.5f * t * (1.f + erff(t * 0.70710678118654752f));
    }
}

__global__ void softmax_rows(float* __restrict__ S, int len)
{
    __shared__ float red[256];
    const int tid = threadIdx.x;
    float* p = S + (size_t)blockIdx.x * len;
    float mx = -3.4e38f;
    for (int i = tid; i < len; i += 256) mx = fmaxf(mx, p[i]);
    red[tid] = mx; __syncthreads();
    for (int st = 128; st > 0; st >>= 1) { if (tid < st) red[tid] = fmaxf(red[tid], red[tid + st]); __syncthreads(); }
    mx = red[0]; __syncthreads();
    float sum = 0.f;
    for (int i = tid; i < len; i += 256) { const float e = expf(p[i] - mx); p[i] = e; sum += e; }
    red[tid] = sum; __syncthreads();
    for (int st = 128; st > 0; st >>= 1) { if (tid < st) red[tid] += red[tid + st]; __syncthreads(); }
    const float inv = 1.f / red[0];
    for (int i = tid; i < len; i += 256) p[i] *= inv;
}

__global__ void bcast_rows(const float* __restrict__ src, float* __restrict__ dst, int rowelems)
{
    const int idx = blockIdx.x * blockDim.x + threadIdx.x;
    const int total = NB * rowelems;
    if (idx >= total) return;
    dst[idx] = src[idx % rowelems];
}

__global__ void concat2(const float* __restrict__ a, const float* __restrict__ b,
                        float* __restrict__ out, int rows)
{
    const int idx = blockIdx.x * blockDim.x + threadIdx.x;
    if (idx >= rows * 2 * FDIM) return;
    const int c = idx % (2 * FDIM);
    const int r = idx / (2 * FDIM);
    out[idx] = (c < FDIM) ? a[(size_t)r * FDIM + c] : b[(size_t)r * FDIM + c - FDIM];
}

__global__ void gate_combine(const float* __restrict__ g, const float* __restrict__ mo,
                             float* __restrict__ cf, int n)
{
    const int idx = blockIdx.x * blockDim.x + threadIdx.x;
    if (idx >= n) return;
    const float gg = g[idx];
    cf[idx] = gg * cf[idx] + (1.f - gg) * mo[idx];
}

// softmax over axis=1 of logits[B, N, C] (over N for each (b,c))
__global__ void softmax_axis1(const float* __restrict__ logits, float* __restrict__ rw)
{
    const int idx = blockIdx.x * blockDim.x + threadIdx.x;
    if (idx >= NB * NCON) return;
    const int c = idx % NCON, b = idx / NCON;
    const float* lp = logits + (size_t)b * NCON * NCON + c;
    float mx = -3.4e38f;
    for (int n = 0; n < NCON; ++n) mx = fmaxf(mx, lp[(size_t)n * NCON]);
    float sum = 0.f;
    for (int n = 0; n < NCON; ++n) sum += expf(lp[(size_t)n * NCON] - mx);
    const float inv = 1.f / sum;
    float* rp = rw + (size_t)b * NCON * NCON + c;
    for (int n = 0; n < NCON; ++n) rp[(size_t)n * NCON] = expf(lp[(size_t)n * NCON] - mx) * inv;
}

// wf[b,c,d] = sum_n rw[b,n,c] * cf[b,n,d]
__global__ void weighted_feats(const float* __restrict__ rw, const float* __restrict__ cf,
                               float* __restrict__ wf)
{
    const int idx = blockIdx.x * blockDim.x + threadIdx.x;
    if (idx >= NB * NCON * FDIM) return;
    const int d = idx % FDIM;
    const int c = (idx / FDIM) % NCON;
    const int b = idx / (FDIM * NCON);
    const float* rp = rw + (size_t)b * NCON * NCON + c;
    const float* cp = cf + (size_t)b * NCON * FDIM + d;
    float s = 0.f;
    for (int n = 0; n < NCON; ++n) s += rp[(size_t)n * NCON] * cp[(size_t)n * FDIM];
    wf[idx] = s;
}

__global__ void concepts_norm(const float* __restrict__ wf, float* __restrict__ concepts,
                              float* __restrict__ conf, int final_it)
{
    const int idx = blockIdx.x * blockDim.x + threadIdx.x;
    if (idx >= NB * NCON) return;
    const float* wp = wf + (size_t)idx * FDIM;
    float* op = concepts + (size_t)idx * FDIM;
    float ss = 0.f;
    for (int d = 0; d < FDIM; ++d) ss += wp[d] * wp[d];
    const float nrm = sqrtf(ss);
    const float s = nrm / (1.f + nrm * nrm);
    for (int d = 0; d < FDIM; ++d) op[d] = wp[d] * s;
    if (final_it) conf[idx] = 1.f / (1.f + expf(-(nrm * s)));
}

// logits[b,n,c] += dot(concepts[b,c], cf[b,n])
__global__ void logits_update(float* __restrict__ logits, const float* __restrict__ concepts,
                              const float* __restrict__ cf)
{
    const int idx = blockIdx.x * blockDim.x + threadIdx.x;
    if (idx >= NB * NCON * NCON) return;
    const int c = idx % NCON;
    const int n = (idx / NCON) % NCON;
    const int b = idx / (NCON * NCON);
    const float* cp = concepts + ((size_t)b * NCON + c) * FDIM;
    const float* fp = cf + ((size_t)b * NCON + n) * FDIM;
    float s = 0.f;
    for (int d = 0; d < FDIM; ++d) s += cp[d] * fp[d];
    logits[idx] += s;
}

__global__ void sigmoid_pos(const float* __restrict__ bbox, float* __restrict__ pos)
{
    const int idx = blockIdx.x * blockDim.x + threadIdx.x;
    if (idx >= NB * NCON * 2) return;
    const int k = idx & 1;
    const int bn = idx >> 1;
    pos[idx] = 1.f / (1.f + expf(-bbox[bn * 4 + k]));
}

// ---------------------------------------------------------------------------
// Host orchestration
// ---------------------------------------------------------------------------
static inline unsigned cdiv_u(unsigned a, unsigned b) { return (a + b - 1) / b; }

extern "C" void kernel_launch(void* const* d_in, const int* in_sizes, int n_in,
                              void* d_out, int out_size, void* d_ws, size_t ws_size,
                              hipStream_t stream)
{
    (void)in_sizes; (void)n_in; (void)out_size; (void)ws_size;

    // JAX pytree flatten order: dict keys sorted alphabetically, depth-first.
    enum {
        IN_F0 = 0, IN_F1, IN_F2,
        IN_A_BK, IN_A_BO, IN_A_BQ, IN_A_BV, IN_A_WK, IN_A_WO, IN_A_WQ, IN_A_WV,
        IN_BBOX_B1, IN_BBOX_B2, IN_BBOX_W1, IN_BBOX_W2,
        IN_COLOR_B, IN_COLOR_W,
        IN_LAT_B0, IN_LAT_B1, IN_LAT_B2, IN_LAT_W0, IN_LAT_W1, IN_LAT_W2,
        IN_OUT_B0, IN_OUT_B1, IN_OUT_B2, IN_OUT_W0, IN_OUT_W1, IN_OUT_W2,
        IN_GATE_B, IN_GATE_W, IN_LN_B, IN_LN_G,
        IN_MAT_B, IN_MAT_W,
        IN_M_BK, IN_M_BO, IN_M_BQ, IN_M_BV, IN_M_WK, IN_M_WO, IN_M_WQ, IN_M_WV,
        IN_MEMORY,
        IN_OBJ_B1, IN_OBJ_B2, IN_OBJ_W1, IN_OBJ_W2,
        IN_PROJ_B, IN_PROJ_W, IN_QUERIES,
        IN_REL_B1, IN_REL_B2, IN_REL_B3, IN_REL_W1, IN_REL_W2, IN_REL_W3,
        IN_ROUTE_B, IN_ROUTE_W,
        IN_SHAPE_B, IN_SHAPE_W, IN_SIZE_B, IN_SIZE_W
    };
    auto F = [&](int i) { return (const float*)d_in[i]; };

    // workspace bump allocator
    float* wsf = (float*)d_ws;
    size_t off = 0;
    auto alloc = [&](size_t n) { float* p = wsf + off; off += n; return p; };

    float* lat0 = alloc((size_t)NB * 256 * 784);
    float* lat1 = alloc((size_t)NB * 256 * 196);
    float* lat2 = alloc((size_t)NB * 256 * 49);
    float* p1   = alloc((size_t)NB * 256 * 196);
    float* p0   = alloc((size_t)NB * 256 * 784);
    float* c0t  = alloc((size_t)NB * 784 * 256);   // token-major conv outputs
    float* c1t  = alloc((size_t)NB * 196 * 256);
    float* c2t  = alloc((size_t)NB * 49 * 256);
    float* x0   = alloc((size_t)NB * 784 * 256);
    float* xp   = alloc((size_t)NB * 784 * FDIM);
    float* xg   = alloc((size_t)NB * 784 * FDIM);
    float* qin  = alloc((size_t)NB * NCON * FDIM);
    float* qp   = alloc((size_t)NB * NCON * FDIM);
    float* kp   = alloc((size_t)NB * 784 * FDIM);
    float* vp   = alloc((size_t)NB * 784 * FDIM);
    float* S1   = alloc((size_t)NB * NHEAD * NCON * 784);
    float* o1   = alloc((size_t)NB * NCON * FDIM);
    float* cf   = alloc((size_t)NB * NCON * FDIM);
    float* memin= alloc((size_t)NB * NCON * FDIM);
    float* q2   = alloc((size_t)NB * NCON * FDIM);
    float* k2   = alloc((size_t)NB * NCON * FDIM);
    float* v2   = alloc((size_t)NB * NCON * FDIM);
    float* S2   = alloc((size_t)NB * NHEAD * NCON * NCON);
    float* o2   = alloc((size_t)NB * NCON * FDIM);
    float* mo   = alloc((size_t)NB * NCON * FDIM);
    float* cat  = alloc((size_t)NB * NCON * 2 * FDIM);
    float* gbuf = alloc((size_t)NB * NCON * FDIM);
    float* logits = alloc((size_t)NB * NCON * NCON);
    float* rw     = alloc((size_t)NB * NCON * NCON);
    float* wfb    = alloc((size_t)NB * NCON * FDIM);
    float* hb     = alloc((size_t)NB * NCON * 256);
    float* hobj   = alloc((size_t)NB * NCON * 128);
    float* pos    = alloc((size_t)NB * NCON * 2);
    float* h1     = alloc((size_t)NB * NCON * NCON * FDIM); // 82 MB
    float* h2     = alloc((size_t)NB * NCON * NCON * 256);  // 41 MB

    // conv-phase scratch aliases the relation scratch (disjoint lifetimes):
    // im2col buffers (9.5M floats) inside h1 (20.5M); wT (1.8M) inside h2 (10.2M)
    float* col0 = h1;
    float* col1 = col0 + (size_t)NB * 784 * CK;
    float* col2 = col1 + (size_t)NB * 196 * CK;
    float* wt0  = h2;
    float* wt1  = wt0 + (size_t)256 * CK;
    float* wt2  = wt1 + (size_t)256 * CK;

    // output regions (flat concat of the returned tuple)
    float* out = (float*)d_out;
    float* o_conf     = out;                                // [4,100]
    float* o_concepts = o_conf + NB * NCON;                 // [4,100,512]
    float* o_rel      = o_concepts + NB * NCON * FDIM;      // [4,100,100,4]
    float* o_bbox     = o_rel + NB * NCON * NCON * 4;       // [4,100,4]
    float* o_obj      = o_bbox + NB * NCON * 4;             // [4,100,1]
    float* o_attrs    = o_obj + NB * NCON;                  // [4,100,27]

    const int T = 256;
    auto gemm = [&](const float* A, const float* W, const float* bias, float* C,
                    int M, int N, int K, int ldc, int act) {
        dim3 grid(cdiv_u(M, 64), cdiv_u(N, 64));
        gemm_wmma<<<grid, dim3(128), 0, stream>>>(A, W, bias, C, M, N, K, ldc, act);
    };

    // ---- FPN: 1x1 laterals + top-down, then 3x3 convs as im2col GEMMs ----
    conv1x1<<<cdiv_u(NB * 256 * 784, T), T, 0, stream>>>(F(IN_F0), F(IN_LAT_W0), F(IN_LAT_B0), lat0, 40, 784);
    conv1x1<<<cdiv_u(NB * 256 * 196, T), T, 0, stream>>>(F(IN_F1), F(IN_LAT_W1), F(IN_LAT_B1), lat1, 112, 196);
    conv1x1<<<cdiv_u(NB * 256 * 49,  T), T, 0, stream>>>(F(IN_F2), F(IN_LAT_W2), F(IN_LAT_B2), lat2, 320, 49);
    upsample_add<<<cdiv_u(NB * 256 * 196, T), T, 0, stream>>>(lat1, lat2, p1, 14);
    upsample_add<<<cdiv_u(NB * 256 * 784, T), T, 0, stream>>>(lat0, p1, p0, 28);

    im2col3x3<<<cdiv_u(NB * 784 * CK, T), T, 0, stream>>>(p0,   col0, 28);
    im2col3x3<<<cdiv_u(NB * 196 * CK, T), T, 0, stream>>>(p1,   col1, 14);
    im2col3x3<<<cdiv_u(NB * 49 * CK,  T), T, 0, stream>>>(lat2, col2, 7);
    transpose_oc<<<cdiv_u(256 * CK, T), T, 0, stream>>>(F(IN_OUT_W0), wt0, 256, CK);
    transpose_oc<<<cdiv_u(256 * CK, T), T, 0, stream>>>(F(IN_OUT_W1), wt1, 256, CK);
    transpose_oc<<<cdiv_u(256 * CK, T), T, 0, stream>>>(F(IN_OUT_W2), wt2, 256, CK);
    gemm(col0, wt0, F(IN_OUT_B0), c0t, NB * 784, 256, CK, 256, ACT_NONE);
    gemm(col1, wt1, F(IN_OUT_B1), c1t, NB * 196, 256, CK, 256, ACT_NONE);
    gemm(col2, wt2, F(IN_OUT_B2), c2t, NB * 49,  256, CK, 256, ACT_NONE);

    // ---- tokens + projection + LN + GELU ----
    build_tokens<<<cdiv_u(NB * 784 * 256, T), T, 0, stream>>>(c0t, c1t, c2t, x0);
    gemm(x0, F(IN_PROJ_W), F(IN_PROJ_B), xp, NB * 784, FDIM, 256, FDIM, ACT_NONE);
    ln_gelu<<<NB * 784, 256, 0, stream>>>(xp, F(IN_LN_G), F(IN_LN_B), xg);

    // ---- concept cross-attention (queries -> tokens) ----
    bcast_rows<<<cdiv_u(NB * NCON * FDIM, T), T, 0, stream>>>(F(IN_QUERIES), qin, NCON * FDIM);
    gemm(qin, F(IN_A_WQ), F(IN_A_BQ), qp, NB * NCON, FDIM, FDIM, FDIM, ACT_NONE);
    gemm(xg,  F(IN_A_WK), F(IN_A_BK), kp, NB * 784,  FDIM, FDIM, FDIM, ACT_NONE);
    gemm(xg,  F(IN_A_WV), F(IN_A_BV), vp, NB * 784,  FDIM, FDIM, FDIM, ACT_NONE);
    attn_qk_wmma<<<dim3(cdiv_u(NCON, 16), cdiv_u(784, 16), NB * NHEAD), 32, 0, stream>>>(qp, kp, S1, NCON, 784, 0.125f);
    softmax_rows<<<NB * NHEAD * NCON, 256, 0, stream>>>(S1, 784);
    attn_av_wmma<<<dim3(cdiv_u(NCON, 64), NB * NHEAD), 128, 0, stream>>>(S1, vp, o1, NCON, 784);
    gemm(o1, F(IN_A_WO), F(IN_A_BO), cf, NB * NCON, FDIM, FDIM, FDIM, ACT_NONE);

    // ---- memory cross-attention (cf -> memory) ----
    bcast_rows<<<cdiv_u(NB * NCON * FDIM, T), T, 0, stream>>>(F(IN_MEMORY), memin, NCON * FDIM);
    gemm(cf,    F(IN_M_WQ), F(IN_M_BQ), q2, NB * NCON, FDIM, FDIM, FDIM, ACT_NONE);
    gemm(memin, F(IN_M_WK), F(IN_M_BK), k2, NB * NCON, FDIM, FDIM, FDIM, ACT_NONE);
    gemm(memin, F(IN_M_WV), F(IN_M_BV), v2, NB * NCON, FDIM, FDIM, FDIM, ACT_NONE);
    attn_qk_wmma<<<dim3(cdiv_u(NCON, 16), cdiv_u(NCON, 16), NB * NHEAD), 32, 0, stream>>>(q2, k2, S2, NCON, NCON, 0.125f);
    softmax_rows<<<NB * NHEAD * NCON, 256, 0, stream>>>(S2, NCON);
    attn_av_wmma<<<dim3(cdiv_u(NCON, 64), NB * NHEAD), 128, 0, stream>>>(S2, v2, o2, NCON, NCON);
    gemm(o2, F(IN_M_WO), F(IN_M_BO), mo, NB * NCON, FDIM, FDIM, FDIM, ACT_NONE);

    // ---- gating ----
    concat2<<<cdiv_u(NB * NCON * 2 * FDIM, T), T, 0, stream>>>(cf, mo, cat, NB * NCON);
    gemm(cat, F(IN_GATE_W), F(IN_GATE_B), gbuf, NB * NCON, FDIM, 2 * FDIM, FDIM, ACT_SIGM);
    gate_combine<<<cdiv_u(NB * NCON * FDIM, T), T, 0, stream>>>(gbuf, mo, cf, NB * NCON * FDIM);

    // ---- dynamic concept routing (3 iterations) ----
    gemm(cf, F(IN_ROUTE_W), F(IN_ROUTE_B), logits, NB * NCON, NCON, FDIM, NCON, ACT_NONE);
    for (int it = 0; it < 3; ++it) {
        softmax_axis1<<<cdiv_u(NB * NCON, T), T, 0, stream>>>(logits, rw);
        weighted_feats<<<cdiv_u(NB * NCON * FDIM, T), T, 0, stream>>>(rw, cf, wfb);
        concepts_norm<<<cdiv_u(NB * NCON, T), T, 0, stream>>>(wfb, o_concepts, o_conf, it == 2 ? 1 : 0);
        if (it < 2)
            logits_update<<<cdiv_u(NB * NCON * NCON, T), T, 0, stream>>>(logits, o_concepts, cf);
    }

    // ---- heads ----
    gemm(cf, F(IN_BBOX_W1), F(IN_BBOX_B1), hb, NB * NCON, 256, FDIM, 256, ACT_RELU);
    gemm(hb, F(IN_BBOX_W2), F(IN_BBOX_B2), o_bbox, NB * NCON, 4, 256, 4, ACT_NONE);
    sigmoid_pos<<<cdiv_u(NB * NCON * 2, T), T, 0, stream>>>(o_bbox, pos);
    gemm(cf, F(IN_OBJ_W1), F(IN_OBJ_B1), hobj, NB * NCON, 128, FDIM, 128, ACT_RELU);
    gemm(hobj, F(IN_OBJ_W2), F(IN_OBJ_B2), o_obj, NB * NCON, 1, 128, 1, ACT_NONE);
    gemm(cf, F(IN_SIZE_W),  F(IN_SIZE_B),  o_attrs + 0,  NB * NCON, 3,  FDIM, 27, ACT_NONE);
    gemm(cf, F(IN_COLOR_W), F(IN_COLOR_B), o_attrs + 3,  NB * NCON, 11, FDIM, 27, ACT_NONE);
    gemm(cf, F(IN_SHAPE_W), F(IN_SHAPE_B), o_attrs + 14, NB * NCON, 5,  FDIM, 27, ACT_NONE);
    gemm(cf, F(IN_MAT_W),   F(IN_MAT_B),   o_attrs + 19, NB * NCON, 8,  FDIM, 27, ACT_NONE);

    // ---- pairwise relations (fused gather-GEMM + 2 dense GEMMs) ----
    rel1_wmma<<<dim3(NB * NCON * NCON / 64, FDIM / 64), 128, 0, stream>>>(cf, pos, F(IN_REL_W1), F(IN_REL_B1), h1);
    gemm(h1, F(IN_REL_W2), F(IN_REL_B2), h2, NB * NCON * NCON, 256, FDIM, 256, ACT_RELU);
    gemm(h2, F(IN_REL_W3), F(IN_REL_B3), o_rel, NB * NCON * NCON, 4, 256, 4, ACT_NONE);
}
